// MultiOmicsGNN_33071248179789
// MI455X (gfx1250) — matrix operand order repaired and checked by reading, not compiled, not run on hardware
//
#include <hip/hip_runtime.h>

// ---------------------------------------------------------------------------
// MultiOmicsGNN forward for MI455X (gfx1250, wave32, WMMA)
//
//   per omics o:  H1 = relu(bn1(X_o @ W1_o + b1))        [256 x 256]  (split-K WMMA)
//                 Xcat[o] = bn2(H1 @ W2_o + b2)          [256 x 128]
//   GCN on fully-connected graph collapses exactly to:
//                 v1 = relu(colmean(Xcat) @ gW0 + gb0)
//                 v2 = relu(v1 @ gW1 + gb1)              (== mean-pooled g)
//   head:         out = relu(v2 @ hW1 + hb1) @ hW2 + hb2
//
// W is pre-packed once per layer into f16 WMMA-fragment order so the GEMM's
// B fragments are single contiguous 32-byte loads (2x global_load_b128).
// ---------------------------------------------------------------------------

#define NSAMP   256
#define HID     256
#define EMB     128
#define NNODES  768
#define KCHUNK  2048                 // K elements per split-K chunk (64 wmma steps)
#define MAXCHUNKS 14                 // ceil(27008/2048)
#define BN_SCALE 0.9995003746f       // 1/sqrt(1 + 1e-3)

typedef __attribute__((ext_vector_type(16))) _Float16 v16h;
typedef __attribute__((ext_vector_type(8)))  float    v8f;

// Packed-W layout: Wh[ ((kb*N + col)*2 + g)*16 + e ]  holds W[kb*32 + 8g + (e<8 ? e : e+8)][col]
// (kb = K-step, g = half-wave, e = fragment element). Zero-padded to Kp = ceil32(K).

// ---------------------------------------------------------------------------
// Pack W [K, N] f32 (row-major) -> f16 fragment-ordered Wh, zero-padded K tail.
// One thread per (kb, g, col): 16 coalesced strided reads, one 32B packed write.
// ---------------------------------------------------------------------------
__global__ void __launch_bounds__(256)
momics_pack_w_f16(const float* __restrict__ W, _Float16* __restrict__ Wh, int K, int N)
{
    const int tid = (int)(blockIdx.x * blockDim.x + threadIdx.x);
    const int nkb = (K + 31) >> 5;
    if (tid >= nkb * 2 * N) return;

    const int col = tid % N;
    const int g   = (tid / N) & 1;
    const int kb  = tid / (2 * N);
    const int kbase = kb * 32 + 8 * g;

    v16h frag;
    #pragma unroll
    for (int e = 0; e < 16; ++e) {
        const int k = kbase + ((e < 8) ? e : e + 8);
        const float v = (k < K) ? W[(size_t)k * (size_t)N + (size_t)col] : 0.0f;
        frag[e] = (_Float16)v;
    }
    *(v16h*)(Wh + ((size_t)(kb * N + col) * 2 + g) * 16) = frag;
}

// ---------------------------------------------------------------------------
// Split-K GEMM partial: each wave computes a 16x64 strip of C for one K-chunk.
// A (f32) loaded as 4x b128 + cvt_pk; B fragments are contiguous v16h loads
// from the packed Wh. K tail: clamped A addresses x zero-padded B == 0.
// ---------------------------------------------------------------------------
__global__ void __launch_bounds__(128)
momics_gemm_wmma_partial(const float* __restrict__ X,       // [M, K] row-major f32
                         const _Float16* __restrict__ Wh,   // packed [Kp/32][N][2][16]
                         float* __restrict__ partial,       // [nchunks, M, N]
                         int M, int K, int N)
{
    const int lane = threadIdx.x & 31;
    const int wid  = (int)((blockIdx.x * blockDim.x + threadIdx.x) >> 5);

    const int stripsN       = N >> 6;            // 64-wide output strips
    const int wavesPerChunk = stripsN * (M >> 4);

    const int ci  = wid / wavesPerChunk;
    const int rem = wid - ci * wavesPerChunk;
    const int tm  = (rem / stripsN) << 4;        // tile row base
    const int tn  = (rem % stripsN) << 6;        // strip col base

    const int g  = lane >> 4;                    // half-wave (ISA K-split)
    const int mn = lane & 15;                    // A-row / B-col within tile

    const int Kp   = (K + 31) & ~31;
    const int Kal  = K & ~31;
    const int kBeg = ci * KCHUNK;
    int kEndP = kBeg + KCHUNK; if (kEndP > Kp)  kEndP = Kp;   // padded end
    int kFull = kBeg + KCHUNK; if (kFull > Kal) kFull = Kal;  // fully in-bounds end

    const float* __restrict__ Arow = X + (size_t)(tm + mn) * (size_t)K;

    // running pointers (strength-reduced; bumped per K-step)
    const float*    ap = Arow + kBeg + 8 * g;
    const _Float16* bp = Wh + ((size_t)(tn + mn) * 2 + g) * 16
                            + (size_t)(kBeg >> 5) * (size_t)N * 32;
    const size_t bstep = (size_t)N * 32;         // halves per 32-K step

    v8f acc[4];
    acc[0] = (v8f){}; acc[1] = (v8f){}; acc[2] = (v8f){}; acc[3] = (v8f){};

    // ---------------- unguarded, fully vectorized main body ----------------
    #pragma unroll 4
    for (int k0 = kBeg; k0 < kFull; k0 += 32, ap += 32, bp += bstep) {
        const float4 q0 = *(const float4*)(ap +  0);
        const float4 q1 = *(const float4*)(ap +  4);
        const float4 q2 = *(const float4*)(ap + 16);
        const float4 q3 = *(const float4*)(ap + 20);

        __builtin_prefetch(ap + 64, 0, 3);       // speculative: always safe, no guard

        v16h a;
        a[0]  = (_Float16)q0.x; a[1]  = (_Float16)q0.y;
        a[2]  = (_Float16)q0.z; a[3]  = (_Float16)q0.w;
        a[4]  = (_Float16)q1.x; a[5]  = (_Float16)q1.y;
        a[6]  = (_Float16)q1.z; a[7]  = (_Float16)q1.w;
        a[8]  = (_Float16)q2.x; a[9]  = (_Float16)q2.y;
        a[10] = (_Float16)q2.z; a[11] = (_Float16)q2.w;
        a[12] = (_Float16)q3.x; a[13] = (_Float16)q3.y;
        a[14] = (_Float16)q3.z; a[15] = (_Float16)q3.w;

        #pragma unroll
        for (int s = 0; s < 4; ++s) {
            const v16h b = *(const v16h*)(bp + (size_t)(16 * s) * 32);
            acc[s] = __builtin_amdgcn_wmma_f32_16x16x32_f16(
                         false, a, false, b, (short)0, acc[s], false, false);
        }
    }

    // ------- K tail (last chunk only): clamped A x zero-padded B -------
    if (kFull < kEndP) {
        v16h a;
        #pragma unroll
        for (int e = 0; e < 16; ++e) {
            const int k  = kFull + ((e < 8) ? e : e + 8) + 8 * g;
            const int kc = (k < K) ? k : (K - 1);          // in-bounds, finite
            a[e] = (_Float16)Arow[kc];                     // x B==0 in pad region
        }
        #pragma unroll
        for (int s = 0; s < 4; ++s) {
            const v16h b = *(const v16h*)(bp + (size_t)(16 * s) * 32);
            acc[s] = __builtin_amdgcn_wmma_f32_16x16x32_f16(
                         false, a, false, b, (short)0, acc[s], false, false);
        }
    }

    // C/D layout: element r -> row = tm + r + 8*g, col = tn + 16*s + mn
    float* __restrict__ P = partial + (size_t)ci * (size_t)M * (size_t)N;
    #pragma unroll
    for (int s = 0; s < 4; ++s) {
        #pragma unroll
        for (int r = 0; r < 8; ++r) {
            P[(size_t)(tm + r + 8 * g) * (size_t)N + (size_t)(tn + 16 * s + mn)] = acc[s][r];
        }
    }
}

// ---------------------------------------------------------------------------
// Deterministic split-K reduction + bias + inference-BN + optional ReLU.
// ---------------------------------------------------------------------------
__global__ void __launch_bounds__(256)
momics_reduce_bn_act(const float* __restrict__ partial, int nchunks, int M, int N,
                     const float* __restrict__ bias, const float* __restrict__ gamma,
                     const float* __restrict__ beta, float* __restrict__ out, int doRelu)
{
    const int idx = (int)(blockIdx.x * blockDim.x + threadIdx.x);
    if (idx >= M * N) return;
    const int col = idx % N;
    float s = 0.0f;
    for (int c = 0; c < nchunks; ++c)
        s += partial[(size_t)c * (size_t)M * (size_t)N + (size_t)idx];
    s += bias[col];
    s = s * (gamma[col] * BN_SCALE) + beta[col];
    if (doRelu) s = fmaxf(s, 0.0f);
    out[idx] = s;
}

// ---------------------------------------------------------------------------
// GCN (collapsed: uniform degree N => aggregation == column mean) + pool + head.
// ---------------------------------------------------------------------------
__global__ void __launch_bounds__(256)
momics_gcn_head(const float* __restrict__ Xcat,                     // [768, 128]
                const float* __restrict__ gW0, const float* __restrict__ gb0,
                const float* __restrict__ gW1, const float* __restrict__ gb1,
                const float* __restrict__ hW1, const float* __restrict__ hb1,
                const float* __restrict__ hW2, const float* __restrict__ hb2,
                float* __restrict__ out)
{
    __shared__ float cm[EMB];
    __shared__ float v1[EMB];
    __shared__ float v2[EMB];
    __shared__ float red[256];
    const int t = (int)threadIdx.x;

    if (t < EMB) {                       // column mean == GCN aggregation (norm 1/N)
        float s = 0.0f;
        for (int r = 0; r < NNODES; ++r) s += Xcat[r * EMB + t];
        cm[t] = s * (1.0f / (float)NNODES);
    }
    __syncthreads();
    if (t < EMB) {                       // GCN layer 0
        float s = gb0[t];
        for (int k = 0; k < EMB; ++k) s += cm[k] * gW0[k * EMB + t];
        v1[t] = fmaxf(s, 0.0f);
    }
    __syncthreads();
    if (t < EMB) {                       // GCN layer 1 (rows identical -> mean = id)
        float s = gb1[t];
        for (int k = 0; k < EMB; ++k) s += v1[k] * gW1[k * EMB + t];
        v2[t] = fmaxf(s, 0.0f);
    }
    __syncthreads();
    float hval = hb1[t];                 // head layer 1
    for (int k = 0; k < EMB; ++k) hval += v2[k] * hW1[k * HID + t];
    hval = fmaxf(hval, 0.0f);

    red[t] = hval * hW2[t];              // head layer 2 (dot product)
    __syncthreads();
    for (int sft = 128; sft > 0; sft >>= 1) {
        if (t < sft) red[t] += red[t + sft];
        __syncthreads();
    }
    if (t == 0) out[0] = red[0] + hb2[0];
}

// ---------------------------------------------------------------------------
// Host launcher.  d_in order (setup_inputs dict order, flattened in order):
//   0..2: x_rna, x_meth, x_mirna | 3+8*o: enc[o] {W1,b1,g1,be1,W2,b2,g2,be2}
//   27..30: gcn {W,b}x2 | 31..34: head {W1,b1,W2,b2} | 35..36: src,dst (unused)
// ---------------------------------------------------------------------------
extern "C" void kernel_launch(void* const* d_in, const int* in_sizes, int n_in,
                              void* d_out, int out_size, void* d_ws, size_t ws_size,
                              hipStream_t stream)
{
    (void)in_sizes; (void)n_in; (void)out_size; (void)ws_size;

    const float* xs[3]   = {(const float*)d_in[0], (const float*)d_in[1], (const float*)d_in[2]};
    const int    dims[3] = {20000, 27000, 2000};

    float* ws      = (float*)d_ws;
    float* partial = ws;                                         // MAXCHUNKS*256*256 f32
    float* H1      = partial + (size_t)MAXCHUNKS * NSAMP * HID;  // 256*256
    float* Xcat    = H1 + (size_t)NSAMP * HID;                   // 768*128
    _Float16* Wh   = (_Float16*)(Xcat + (size_t)NNODES * EMB);   // <= 27008*256 halves

    for (int o = 0; o < 3; ++o) {
        const float* W1  = (const float*)d_in[3 + 8 * o + 0];
        const float* b1  = (const float*)d_in[3 + 8 * o + 1];
        const float* g1  = (const float*)d_in[3 + 8 * o + 2];
        const float* be1 = (const float*)d_in[3 + 8 * o + 3];
        const float* W2  = (const float*)d_in[3 + 8 * o + 4];
        const float* b2  = (const float*)d_in[3 + 8 * o + 5];
        const float* g2  = (const float*)d_in[3 + 8 * o + 6];
        const float* be2 = (const float*)d_in[3 + 8 * o + 7];
        const int D    = dims[o];
        const int Dp   = (D + 31) & ~31;
        const int nch1 = (Dp + KCHUNK - 1) / KCHUNK;

        // layer 1: H1 = relu(bn1(X @ W1 + b1))  [256 x 256]
        {
            const int packThreads = (Dp >> 5) * 2 * HID;
            momics_pack_w_f16<<<(packThreads + 255) / 256, 256, 0, stream>>>(W1, Wh, D, HID);
            const int waves = nch1 * (NSAMP / 16) * (HID / 64);
            momics_gemm_wmma_partial<<<waves / 4, 128, 0, stream>>>(
                xs[o], Wh, partial, NSAMP, D, HID);
            momics_reduce_bn_act<<<(NSAMP * HID) / 256, 256, 0, stream>>>(
                partial, nch1, NSAMP, HID, b1, g1, be1, H1, 1);
        }
        // layer 2: Xcat[o] = bn2(H1 @ W2 + b2)  [256 x 128]
        {
            const int packThreads = (HID >> 5) * 2 * EMB;
            momics_pack_w_f16<<<(packThreads + 255) / 256, 256, 0, stream>>>(W2, Wh, HID, EMB);
            const int waves = (NSAMP / 16) * (EMB / 64);
            momics_gemm_wmma_partial<<<waves / 4, 128, 0, stream>>>(
                H1, Wh, partial, NSAMP, HID, EMB);
            momics_reduce_bn_act<<<(NSAMP * EMB) / 256, 256, 0, stream>>>(
                partial, 1, NSAMP, EMB, b2, g2, be2, Xcat + (size_t)o * NSAMP * EMB, 0);
        }
    }

    const float* gW0 = (const float*)d_in[27];
    const float* gb0 = (const float*)d_in[28];
    const float* gW1 = (const float*)d_in[29];
    const float* gb1 = (const float*)d_in[30];
    const float* hW1 = (const float*)d_in[31];
    const float* hb1 = (const float*)d_in[32];
    const float* hW2 = (const float*)d_in[33];
    const float* hb2 = (const float*)d_in[34];

    momics_gcn_head<<<1, 256, 0, stream>>>(Xcat, gW0, gb0, gW1, gb1,
                                           hW1, hb1, hW2, hb2, (float*)d_out);
}
